// CoordConv_29180007809227
// MI455X (gfx1250) — matrix operand order repaired and checked by reading, not compiled, not run on hardware
//
#include <hip/hip_runtime.h>
#include <hip/hip_bf16.h>

#define N_NODES 50000
#define N_EDGES 800000

typedef __attribute__((ext_vector_type(16))) _Float16 v16h;
typedef __attribute__((ext_vector_type(8)))  float    v8f;

__device__ __forceinline__ float eluf(float x) { return x > 0.f ? x : __expf(x) - 1.f; }

// ---- WMMA fragment helpers (16x16x32 f16, wave32; layouts per CDNA5 ISA 7.12.2) ----

// A-matrix 16x32 f16: lanes 0-15 hold M=0..15; VGPR v: K offset pattern
__device__ __forceinline__ int a_koff(int v, int lane) {
  return ((v & 4) << 2) | ((lane & 16) >> 1) | ((v & 3) << 1);
}

__device__ __forceinline__ v16h load_a_f16(const _Float16* base, int ld, int lane) {
  v16h a;
  const _Float16* row = base + (lane & 15) * ld;
#pragma unroll
  for (int v = 0; v < 8; ++v) {
    int k = a_koff(v, lane);
    a[2 * v]     = row[k];
    a[2 * v + 1] = row[k + 1];
  }
  return a;
}

__device__ __forceinline__ v16h load_a_f32(const float* base, int ld, int lane) {
  v16h a;
  const float* row = base + (lane & 15) * ld;
#pragma unroll
  for (int v = 0; v < 8; ++v) {
    int k = a_koff(v, lane);
    a[2 * v]     = (_Float16)row[k];
    a[2 * v + 1] = (_Float16)row[k + 1];
  }
  return a;
}

// B fragments are pre-swizzled in workspace by k_cvt_swz:
//   frag element j of lane L for chunk (kc,nt) = W[(kc*32 + ((j&8)<<1) + ((L&16)>>1) + (j&7))][nt*16 + (L&15)]
// stored contiguously: dst[(((kc*NT + nt)*32 + L)*16 + j]  -> one aligned 32B load per lane.
__device__ __forceinline__ v16h load_b_frag(const _Float16* Wf, int NT, int kc, int nt, int lane) {
  const _Float16* p = Wf + (unsigned)(((kc * NT + nt) * 32 + lane) * 16);
  return *(const v16h*)p;
}

__device__ __forceinline__ v8f wmma16(v16h a, v16h b, v8f c) {
  return __builtin_amdgcn_wmma_f32_16x16x32_f16(false, a, false, b, (short)0, c, false, false);
}

// C/D layout: lane l, reg r -> row m = r + ((l&16)>>1), col n = l&15.
__device__ __forceinline__ int c_row(int r, int lane) { return r + ((lane & 16) >> 1); }

// ---------------- utility kernels ----------------

__global__ void k_zero(float* p, int n) {
  int i = blockIdx.x * blockDim.x + threadIdx.x;
  if (i < n) p[i] = 0.f;
}

// Convert f32 weights [K][Nc] row-major -> f16 B-fragment layout (see load_b_frag).
__global__ void k_cvt_swz(const float* __restrict__ W, _Float16* __restrict__ dst,
                          int K, int Nc) {
  int i = blockIdx.x * blockDim.x + threadIdx.x;
  if (i >= K * Nc) return;
  int NT = Nc >> 4;
  int j    = i & 15;
  int lane = (i >> 4) & 31;
  int frag = i >> 9;
  int nt = frag % NT;
  int kc = frag / NT;
  int k = kc * 32 + ((j & 8) << 1) + ((lane & 16) >> 1) + (j & 7);
  int n = nt * 16 + (lane & 15);
  dst[i] = (_Float16)W[(unsigned)(k * Nc + n)];
}

// ---------------- edge softmax passes ----------------

__global__ void k_edge_score(const float* __restrict__ off, const int* __restrict__ dst,
                             float* __restrict__ smax) {
  int e = blockIdx.x * blockDim.x + threadIdx.x;
  if (e >= N_EDGES) return;
  float s = 1.f / (fabsf(off[2 * e]) + fabsf(off[2 * e + 1]) + 0.001f);
  // score > 0 always, so int-bit compare == float compare
  atomicMax((int*)(smax + dst[e]), __float_as_int(s));
}

__global__ void k_edge_exp(const float* __restrict__ off, const int* __restrict__ dst,
                           const float* __restrict__ smax, float* __restrict__ ex,
                           float* __restrict__ denom) {
  int e = blockIdx.x * blockDim.x + threadIdx.x;
  if (e >= N_EDGES) return;
  float s = 1.f / (fabsf(off[2 * e]) + fabsf(off[2 * e + 1]) + 0.001f);
  float v = __expf(s - smax[dst[e]]);
  ex[e] = v;
  atomicAdd(denom + dst[e], v);
}

// ---------------- fused edge MLP + message scatter ----------------
// One wave handles 16 edges. Layer1 (K=2) on VALU, layer2 (128->64) on WMMA.
__global__ __launch_bounds__(256) void k_edge_msg(
    const float* __restrict__ off, const int* __restrict__ src, const int* __restrict__ dst,
    const float* __restrict__ feat, const float* __restrict__ kW1, const float* __restrict__ kb1,
    const _Float16* __restrict__ kW2f, const float* __restrict__ kb2,
    const float* __restrict__ ex, const float* __restrict__ denom, float* __restrict__ agg) {
  __shared__ _Float16 hid[8][16][128];
  __shared__ int   sS[8][16];
  __shared__ int   sD[8][16];
  __shared__ float sWt[8][16];

  int lane = threadIdx.x & 31;
  int wave = threadIdx.x >> 5;
  unsigned tile = blockIdx.x * 8 + wave;  // grid is exact: 6250*8 == 50000 tiles
  unsigned e0 = tile * 16;

  if (lane < 16) {
    unsigned e = e0 + lane;
    int s = src[e];
    int d = dst[e];
    sS[wave][lane] = s;
    sD[wave][lane] = d;
    sWt[wave][lane] = ex[e] / denom[d];
    __builtin_prefetch(feat + (unsigned)s * 64u, 0, 1);
  }

  { // hidden = elu(offsets @ kW1 + kb1), 2 lanes per edge
    int el = lane >> 1, half = lane & 1;
    float o0 = off[(e0 + el) * 2 + 0];
    float o1 = off[(e0 + el) * 2 + 1];
#pragma unroll
    for (int j = 0; j < 64; ++j) {
      int h = half * 64 + j;
      float v = eluf(o0 * kW1[h] + o1 * kW1[128 + h] + kb1[h]);
      hid[wave][el][h] = (_Float16)v;
    }
  }
  __syncthreads();

  // hoist per-edge meta into registers (one LDS read each)
  unsigned sB[8], dB[8];
  float wv[8];
#pragma unroll
  for (int r = 0; r < 8; ++r) {
    int m = c_row(r, lane);
    sB[r] = (unsigned)sS[wave][m] * 64u;
    dB[r] = (unsigned)sD[wave][m] * 64u;
    wv[r] = sWt[wave][m];
  }

  int n = lane & 15;
#pragma unroll
  for (int nt = 0; nt < 4; ++nt) {
    float bias = kb2[nt * 16 + n];
    v8f acc;
#pragma unroll
    for (int r = 0; r < 8; ++r) acc[r] = bias;
#pragma unroll
    for (int kc = 0; kc < 4; ++kc) {
      v16h a = load_a_f16(&hid[wave][0][kc * 32], 128, lane);
      v16h b = load_b_frag(kW2f, 4, kc, nt, lane);
      acc = wmma16(a, b, acc);
    }
    unsigned col = (unsigned)(nt * 16 + n);
#pragma unroll
    for (int r = 0; r < 8; ++r) {
      float val = acc[r] * wv[r] * feat[sB[r] + col];
      atomicAdd(agg + (dB[r] + col), val);
    }
  }
}

// ---------------- self branch: h_self = elu(feat@sW1+b1)@sW2+b2 -> f16 ----------------
__global__ __launch_bounds__(256) void k_self(
    const float* __restrict__ feat, const _Float16* __restrict__ sW1f,
    const float* __restrict__ sb1, const _Float16* __restrict__ sW2f,
    const float* __restrict__ sb2, _Float16* __restrict__ hself) {
  __shared__ _Float16 hid[8][16][128];
  int lane = threadIdx.x & 31;
  int wave = threadIdx.x >> 5;
  unsigned tile = blockIdx.x * 8 + wave;
  if (tile >= N_NODES / 16) tile = N_NODES / 16 - 1;  // duplicate tile, same values
  unsigned r0 = tile * 16;
  int n = lane & 15;

#pragma unroll
  for (int nt = 0; nt < 8; ++nt) {
    float bias = sb1[nt * 16 + n];
    v8f acc;
#pragma unroll
    for (int r = 0; r < 8; ++r) acc[r] = bias;
#pragma unroll
    for (int kc = 0; kc < 2; ++kc) {
      v16h a = load_a_f32(feat + r0 * 64u + kc * 32, 64, lane);
      v16h b = load_b_frag(sW1f, 8, kc, nt, lane);
      acc = wmma16(a, b, acc);
    }
#pragma unroll
    for (int r = 0; r < 8; ++r)
      hid[wave][c_row(r, lane)][nt * 16 + n] = (_Float16)eluf(acc[r]);
  }
  __syncthreads();

#pragma unroll
  for (int nt = 0; nt < 8; ++nt) {
    float bias = sb2[nt * 16 + n];
    v8f acc;
#pragma unroll
    for (int r = 0; r < 8; ++r) acc[r] = bias;
#pragma unroll
    for (int kc = 0; kc < 4; ++kc) {
      v16h a = load_a_f16(&hid[wave][0][kc * 32], 128, lane);
      v16h b = load_b_frag(sW2f, 8, kc, nt, lane);
      acc = wmma16(a, b, acc);
    }
#pragma unroll
    for (int r = 0; r < 8; ++r)
      hself[(r0 + c_row(r, lane)) * 128u + (unsigned)(nt * 16 + n)] = (_Float16)acc[r];
  }
}

// ---------------- MLP head: concat(agg, h_self) -> 4 layers -> out ----------------
__global__ __launch_bounds__(128) void k_head(
    const float* __restrict__ agg, const _Float16* __restrict__ hself,
    const _Float16* __restrict__ mW1f, const float* __restrict__ mb1,
    const _Float16* __restrict__ mW2f, const float* __restrict__ mb2,
    const _Float16* __restrict__ mW3f, const float* __restrict__ mb3,
    const _Float16* __restrict__ mW4f, const float* __restrict__ mb4,
    float* __restrict__ out) {
  __shared__ _Float16 hidA[4][16][128];
  __shared__ _Float16 hidB[4][16][128];
  int lane = threadIdx.x & 31;
  int wave = threadIdx.x >> 5;
  unsigned tile = blockIdx.x * 4 + wave;
  if (tile >= N_NODES / 16) tile = N_NODES / 16 - 1;
  unsigned r0 = tile * 16;
  int n = lane & 15;

  // L1: [agg(64) | hself(128)] @ mW1 + mb1, elu -> hidA
#pragma unroll
  for (int nt = 0; nt < 8; ++nt) {
    float bias = mb1[nt * 16 + n];
    v8f acc;
#pragma unroll
    for (int r = 0; r < 8; ++r) acc[r] = bias;
#pragma unroll
    for (int kc = 0; kc < 6; ++kc) {
      v16h a = (kc < 2) ? load_a_f32(agg + r0 * 64u + kc * 32, 64, lane)
                        : load_a_f16(hself + r0 * 128u + (kc - 2) * 32, 128, lane);
      v16h b = load_b_frag(mW1f, 8, kc, nt, lane);
      acc = wmma16(a, b, acc);
    }
#pragma unroll
    for (int r = 0; r < 8; ++r)
      hidA[wave][c_row(r, lane)][nt * 16 + n] = (_Float16)eluf(acc[r]);
  }
  __syncthreads();

  // L2: hidA -> hidB
#pragma unroll
  for (int nt = 0; nt < 8; ++nt) {
    float bias = mb2[nt * 16 + n];
    v8f acc;
#pragma unroll
    for (int r = 0; r < 8; ++r) acc[r] = bias;
#pragma unroll
    for (int kc = 0; kc < 4; ++kc) {
      v16h a = load_a_f16(&hidA[wave][0][kc * 32], 128, lane);
      v16h b = load_b_frag(mW2f, 8, kc, nt, lane);
      acc = wmma16(a, b, acc);
    }
#pragma unroll
    for (int r = 0; r < 8; ++r)
      hidB[wave][c_row(r, lane)][nt * 16 + n] = (_Float16)eluf(acc[r]);
  }
  __syncthreads();

  // L3: hidB -> hidA
#pragma unroll
  for (int nt = 0; nt < 8; ++nt) {
    float bias = mb3[nt * 16 + n];
    v8f acc;
#pragma unroll
    for (int r = 0; r < 8; ++r) acc[r] = bias;
#pragma unroll
    for (int kc = 0; kc < 4; ++kc) {
      v16h a = load_a_f16(&hidB[wave][0][kc * 32], 128, lane);
      v16h b = load_b_frag(mW3f, 8, kc, nt, lane);
      acc = wmma16(a, b, acc);
    }
#pragma unroll
    for (int r = 0; r < 8; ++r)
      hidA[wave][c_row(r, lane)][nt * 16 + n] = (_Float16)eluf(acc[r]);
  }
  __syncthreads();

  // L4: hidA -> out (no elu)
#pragma unroll
  for (int nt = 0; nt < 8; ++nt) {
    float bias = mb4[nt * 16 + n];
    v8f acc;
#pragma unroll
    for (int r = 0; r < 8; ++r) acc[r] = bias;
#pragma unroll
    for (int kc = 0; kc < 4; ++kc) {
      v16h a = load_a_f16(&hidA[wave][0][kc * 32], 128, lane);
      v16h b = load_b_frag(mW4f, 8, kc, nt, lane);
      acc = wmma16(a, b, acc);
    }
#pragma unroll
    for (int r = 0; r < 8; ++r)
      out[(r0 + c_row(r, lane)) * 128u + (unsigned)(nt * 16 + n)] = acc[r];
  }
}

// ---------------- host launch ----------------

extern "C" void kernel_launch(void* const* d_in, const int* in_sizes, int n_in,
                              void* d_out, int out_size, void* d_ws, size_t ws_size,
                              hipStream_t stream) {
  const float* feat    = (const float*)d_in[0];
  const float* offsets = (const float*)d_in[1];
  const int*   src     = (const int*)d_in[2];
  const int*   dst     = (const int*)d_in[3];
  const float* kW1 = (const float*)d_in[4];
  const float* kb1 = (const float*)d_in[5];
  const float* kW2 = (const float*)d_in[6];
  const float* kb2 = (const float*)d_in[7];
  const float* sW1 = (const float*)d_in[8];
  const float* sb1 = (const float*)d_in[9];
  const float* sW2 = (const float*)d_in[10];
  const float* sb2 = (const float*)d_in[11];
  const float* mW1 = (const float*)d_in[12];
  const float* mb1 = (const float*)d_in[13];
  const float* mW2 = (const float*)d_in[14];
  const float* mb2 = (const float*)d_in[15];
  const float* mW3 = (const float*)d_in[16];
  const float* mb3 = (const float*)d_in[17];
  const float* mW4 = (const float*)d_in[18];
  const float* mb4 = (const float*)d_in[19];
  float* out = (float*)d_out;

  // workspace layout (all f16 regions 32B-aligned by construction)
  float* agg   = (float*)d_ws;                      // N*64
  float* denom = agg + (size_t)N_NODES * 64;        // N
  float* smax  = denom + N_NODES;                   // N
  float* ex    = smax + N_NODES;                    // E
  _Float16* hself = (_Float16*)(ex + N_EDGES);      // N*128 f16
  _Float16* wh    = hself + (size_t)N_NODES * 128;  // pre-swizzled f16 weight fragments
  _Float16* kW2f = wh;
  _Float16* sW1f = kW2f + 128 * 64;
  _Float16* sW2f = sW1f + 64 * 128;
  _Float16* mW1f = sW2f + 128 * 128;
  _Float16* mW2f = mW1f + 192 * 128;
  _Float16* mW3f = mW2f + 128 * 128;
  _Float16* mW4f = mW3f + 128 * 128;

  int zn = N_NODES * 64 + 2 * N_NODES;  // agg + denom + smax (contiguous)
  k_zero<<<(zn + 255) / 256, 256, 0, stream>>>(agg, zn);

  k_cvt_swz<<<(128 * 64 + 255) / 256, 256, 0, stream>>>(kW2, kW2f, 128, 64);
  k_cvt_swz<<<(64 * 128 + 255) / 256, 256, 0, stream>>>(sW1, sW1f, 64, 128);
  k_cvt_swz<<<(128 * 128 + 255) / 256, 256, 0, stream>>>(sW2, sW2f, 128, 128);
  k_cvt_swz<<<(192 * 128 + 255) / 256, 256, 0, stream>>>(mW1, mW1f, 192, 128);
  k_cvt_swz<<<(128 * 128 + 255) / 256, 256, 0, stream>>>(mW2, mW2f, 128, 128);
  k_cvt_swz<<<(128 * 128 + 255) / 256, 256, 0, stream>>>(mW3, mW3f, 128, 128);
  k_cvt_swz<<<(128 * 128 + 255) / 256, 256, 0, stream>>>(mW4, mW4f, 128, 128);

  k_edge_score<<<(N_EDGES + 255) / 256, 256, 0, stream>>>(offsets, dst, smax);
  k_edge_exp<<<(N_EDGES + 255) / 256, 256, 0, stream>>>(offsets, dst, smax, ex, denom);

  k_edge_msg<<<N_EDGES / (16 * 8), 256, 0, stream>>>(offsets, src, dst, feat, kW1, kb1,
                                                     kW2f, kb2, ex, denom, agg);

  int ntiles = N_NODES / 16;  // 3125
  k_self<<<(ntiles + 7) / 8, 256, 0, stream>>>(feat, sW1f, sb1, sW2f, sb2, hself);
  k_head<<<(ntiles + 3) / 4, 128, 0, stream>>>(agg, hself, mW1f, mb1, mW2f, mb2,
                                               mW3f, mb3, mW4f, mb4, out);
}
